// BiTreeLSTM_35940286333324
// MI455X (gfx1250) — matrix operand order repaired and checked by reading, not compiled, not run on hardware
//
#include <hip/hip_runtime.h>
#include <math.h>

// ---------------------------------------------------------------------------
// BiTreeLSTM for MI455X (gfx1250).
// One-time f32->f16 operand staging, then wave32 WMMA f32_16x16x32_f16 GEMMs
// reading pure f16 vector fragments (no conversions in the hot loop), plus
// fused elementwise LSTM-cell kernels, level-by-level over the binary tree.
// ---------------------------------------------------------------------------

typedef _Float16 v16h __attribute__((ext_vector_type(16)));
typedef _Float16 v8h  __attribute__((ext_vector_type(8)));
typedef float    v8f  __attribute__((ext_vector_type(8)));

#define MDIM  512
#define INDIM 1024

__device__ __forceinline__ float sigmoidf_(float x) { return 1.0f / (1.0f + __expf(-x)); }

__device__ __forceinline__ v8f wmma_f16(v16h a, v16h b, v8f c) {
  return __builtin_amdgcn_wmma_f32_16x16x32_f16(
      /*neg_a=*/false, a, /*neg_b=*/false, b,
      /*c_mod=*/(short)0, c, /*reuse_a=*/false, /*reuse_b=*/false);
}

// ---------------------------------------------------------------------------
// C[M x N] = A[M x K] * W[N x K]^T (+ bias[N]); A,W in f16, C fp32 (fp32 acc).
// One wave -> 16x64 tile (A fragment reused 4x); 8 waves/block -> 128x64.
// K % 32 == 0, N % 64 == 0; M arbitrary (clamped loads, guarded stores).
// Software-pipelined with a peeled epilogue: the loop body is branch-free
// (unconditional next-step loads -> 4 WMMAs), so the register rotation needs
// no copy blocks and loads for step k+1 overlap the WMMAs of step k.
// ---------------------------------------------------------------------------
__launch_bounds__(256)
__global__ void gemm_nt_wmma_f16(const _Float16* __restrict__ A, const _Float16* __restrict__ W,
                                 const float* __restrict__ bias, float* __restrict__ C,
                                 int M, int N, int K)
{
  const int lane = threadIdx.x & 31;
  const int wave = threadIdx.x >> 5;
  const int rowBase = (blockIdx.x * 8 + wave) * 16;
  if (rowBase >= M) return;                 // wave-uniform: EXEC stays all-ones
  const int colBase = blockIdx.y * 64;
  const int half = lane >> 4;
  const int l16  = lane & 15;

  int arow = rowBase + l16;
  if (arow >= M) arow = M - 1;              // clamp: edge tiles read valid rows
  // A fragment per lane (16-bit A 16x32), same K order as B:
  //   VGPR0-7 <- K[half*16 .. +15]
  const _Float16* Ap  = A + (size_t)arow * K + half * 16;
  // B fragment per lane (32x16): K[half*16 .. +15] for column l16 -> 16
  // contiguous f16 in the row-major weight matrix.
  const _Float16* Bp0 = W + (size_t)(colBase +  0 + l16) * K + half * 16;
  const _Float16* Bp1 = W + (size_t)(colBase + 16 + l16) * K + half * 16;
  const _Float16* Bp2 = W + (size_t)(colBase + 32 + l16) * K + half * 16;
  const _Float16* Bp3 = W + (size_t)(colBase + 48 + l16) * K + half * 16;

  v8f acc[4] = {};

  auto loadA = [&](int k0) -> v16h {
    v8h lo = *(const v8h*)(Ap + k0);        // 16B vector load
    v8h hi = *(const v8h*)(Ap + k0 + 8);    // 16B vector load
    v16h a;
#pragma unroll
    for (int i = 0; i < 8; ++i) { a[i] = lo[i]; a[8 + i] = hi[i]; }
    return a;
  };

  // Prologue loads (K >= 32 always; here K is 512 or 1024).
  v16h a  = loadA(0);
  v16h b0 = *(const v16h*)(Bp0);
  v16h b1 = *(const v16h*)(Bp1);
  v16h b2 = *(const v16h*)(Bp2);
  v16h b3 = *(const v16h*)(Bp3);

  // Branch-free pipelined main loop over K-32 steps.
  for (int k0 = 0; k0 < K - 32; k0 += 32) {
    const int kn = k0 + 32;
    v16h an  = loadA(kn);
    v16h bn0 = *(const v16h*)(Bp0 + kn);
    v16h bn1 = *(const v16h*)(Bp1 + kn);
    v16h bn2 = *(const v16h*)(Bp2 + kn);
    v16h bn3 = *(const v16h*)(Bp3 + kn);
    acc[0] = wmma_f16(a, b0, acc[0]);
    acc[1] = wmma_f16(a, b1, acc[1]);
    acc[2] = wmma_f16(a, b2, acc[2]);
    acc[3] = wmma_f16(a, b3, acc[3]);
    a = an; b0 = bn0; b1 = bn1; b2 = bn2; b3 = bn3;
  }
  // Peeled epilogue: last K-step.
  acc[0] = wmma_f16(a, b0, acc[0]);
  acc[1] = wmma_f16(a, b1, acc[1]);
  acc[2] = wmma_f16(a, b2, acc[2]);
  acc[3] = wmma_f16(a, b3, acc[3]);

  // C/D 16x16 f32 layout: lane half -> M offset 8, VGPR r -> M row r.
  const int mOff = half * 8;
#pragma unroll
  for (int t = 0; t < 4; ++t) {
    const int col = colBase + t * 16 + l16;
    const float bv = bias ? bias[col] : 0.0f;
#pragma unroll
    for (int r = 0; r < 8; ++r) {
      const int row = rowBase + mOff + r;
      if (row < M) C[(size_t)row * N + col] = acc[t][r] + bv;
    }
  }
}

// ---------------------------------------------------------------------------
// One-time fp32 -> f16 operand staging (grid-stride).
// ---------------------------------------------------------------------------
__global__ void cvt_f16_kernel(const float* __restrict__ src, _Float16* __restrict__ dst, size_t n)
{
  size_t i = (size_t)blockIdx.x * blockDim.x + threadIdx.x;
  size_t stride = (size_t)gridDim.x * blockDim.x;
  for (; i < n; i += stride) dst[i] = (_Float16)src[i];
}

// ---------------------------------------------------------------------------
// Child-sum leaf level. xg row layout (ifoux): [i|f|o|u]; ious_b: [i|o|u].
// Writes fp32 h/c plus f16 mirror of h (GEMM operand for the next level up).
// ---------------------------------------------------------------------------
__global__ void cs_leaf_kernel(const float* __restrict__ xg, const float* __restrict__ ious_b,
                               float* __restrict__ h, float* __restrict__ c,
                               _Float16* __restrict__ h16, int lo, int nl)
{
  int idx = blockIdx.x * blockDim.x + threadIdx.x;
  if (idx >= nl * MDIM) return;
  int r = idx >> 9, j = idx & (MDIM - 1);
  const float* g = xg + (size_t)r * (4 * MDIM);
  float i = sigmoidf_(g[j]          + ious_b[j]);
  float o = sigmoidf_(g[2*MDIM + j] + ious_b[MDIM + j]);
  float u = tanhf    (g[3*MDIM + j] + ious_b[2*MDIM + j]);
  float cc = i * u;
  float hh = o * tanhf(cc);
  size_t node = (size_t)(lo + r) * MDIM + j;
  c[node]   = cc;
  h[node]   = hh;
  h16[node] = (_Float16)hh;
}

// hsum16[r] = f16(h[child_lo + 2r] + h[child_lo + 2r + 1])   (GEMM operand only)
__global__ void cs_hsum_kernel(const float* __restrict__ h, _Float16* __restrict__ hsum16,
                               int child_lo, int nl)
{
  int idx = blockIdx.x * blockDim.x + threadIdx.x;
  if (idx >= nl * MDIM) return;
  int r = idx >> 9, j = idx & (MDIM - 1);
  size_t cl = (size_t)(child_lo + 2 * r) * MDIM + j;
  hsum16[(size_t)r * MDIM + j] = (_Float16)(h[cl] + h[cl + MDIM]);
}

// Child-sum internal node combine.
__global__ void cs_combine_kernel(const float* __restrict__ xg, const float* __restrict__ iou,
                                  const float* __restrict__ fproj, const float* __restrict__ ious_b,
                                  const float* __restrict__ fh_b,
                                  float* __restrict__ h, float* __restrict__ c,
                                  _Float16* __restrict__ h16,
                                  int lo, int child_lo, int nl)
{
  int idx = blockIdx.x * blockDim.x + threadIdx.x;
  if (idx >= nl * MDIM) return;
  int r = idx >> 9, j = idx & (MDIM - 1);
  const float* g  = xg  + (size_t)r * (4 * MDIM);
  const float* io = iou + (size_t)r * (3 * MDIM);
  float i = sigmoidf_(g[j]          + io[j]          + ious_b[j]);
  float o = sigmoidf_(g[2*MDIM + j] + io[MDIM + j]   + ious_b[MDIM + j]);
  float u = tanhf    (g[3*MDIM + j] + io[2*MDIM + j] + ious_b[2*MDIM + j]);
  float fx = g[MDIM + j];
  float fb = fh_b[j];
  float fl = sigmoidf_(fproj[(size_t)(2*r)     * MDIM + j] + fb + fx);
  float fr = sigmoidf_(fproj[(size_t)(2*r + 1) * MDIM + j] + fb + fx);
  size_t cl = (size_t)(child_lo + 2 * r) * MDIM + j;
  float cc = i * u + fl * c[cl] + fr * c[cl + MDIM];
  float hh = o * tanhf(cc);
  size_t node = (size_t)(lo + r) * MDIM + j;
  c[node]   = cc;
  h[node]   = hh;
  h16[node] = (_Float16)hh;
}

// Chain (top-down) combine.  xg/iofuh layout: [i|o|f|u]
__global__ void chain_combine_kernel(const float* __restrict__ xg, const float* __restrict__ proj,
                                     const float* __restrict__ iofuh_b,
                                     float* __restrict__ h, float* __restrict__ c,
                                     _Float16* __restrict__ h16,
                                     int lo, int plo, int nl, int has_parent)
{
  int idx = blockIdx.x * blockDim.x + threadIdx.x;
  if (idx >= nl * MDIM) return;
  int r = idx >> 9, j = idx & (MDIM - 1);
  int q = r >> 1;  // parent's local index within its level
  const float* g = xg + (size_t)r * (4 * MDIM);
  float pi = 0.f, po = 0.f, pf = 0.f, pu = 0.f, pc = 0.f;
  if (has_parent) {
    const float* pr = proj + (size_t)q * (4 * MDIM);
    pi = pr[j]; po = pr[MDIM + j]; pf = pr[2*MDIM + j]; pu = pr[3*MDIM + j];
    pc = c[(size_t)(plo + q) * MDIM + j];
  }
  float i = sigmoidf_(g[j]          + pi + iofuh_b[j]);
  float o = sigmoidf_(g[MDIM + j]   + po + iofuh_b[MDIM + j]);
  float f = sigmoidf_(g[2*MDIM + j] + pf + iofuh_b[2*MDIM + j]);
  float u = tanhf    (g[3*MDIM + j] + pu + iofuh_b[3*MDIM + j]);
  float cc = i * u + f * pc;
  float hh = o * tanhf(cc);
  size_t node = (size_t)(lo + r) * MDIM + j;
  c[node]   = cc;
  h[node]   = hh;
  h16[node] = (_Float16)hh;
}

// Column-max stage 1: each (block.y) chunk reduces its row range per column.
__global__ void colmax_partial_kernel(const float* __restrict__ h, float* __restrict__ partial,
                                      int Ntot)
{
  int j = blockIdx.x * blockDim.x + threadIdx.x;
  if (j >= MDIM) return;
  int chunk = (Ntot + gridDim.y - 1) / gridDim.y;
  int r0 = blockIdx.y * chunk;
  int r1 = r0 + chunk; if (r1 > Ntot) r1 = Ntot;
  float m = -3.402823466e+38f;
  for (int r = r0; r < r1; ++r) m = fmaxf(m, h[(size_t)r * MDIM + j]);
  partial[(size_t)blockIdx.y * MDIM + j] = m;
}

// Stage 2 + output assembly: out[0:512]=h_childsum(root), out[512:1024]=colmax.
__global__ void finalize_kernel(const float* __restrict__ h1_root, const float* __restrict__ partial,
                                float* __restrict__ out, int nchunks)
{
  int j = blockIdx.x * blockDim.x + threadIdx.x;
  if (j >= MDIM) return;
  out[j] = h1_root[j];
  float m = partial[j];
  for (int t = 1; t < nchunks; ++t) m = fmaxf(m, partial[(size_t)t * MDIM + j]);
  out[MDIM + j] = m;
}

// ---------------------------------------------------------------------------
extern "C" void kernel_launch(void* const* d_in, const int* in_sizes, int n_in,
                              void* d_out, int out_size, void* d_ws, size_t ws_size,
                              hipStream_t stream)
{
  const float* inputs  = (const float*)d_in[0];
  const float* ifoux_w = (const float*)d_in[1];
  const float* ifoux_b = (const float*)d_in[2];
  const float* ious_w  = (const float*)d_in[3];
  const float* ious_b  = (const float*)d_in[4];
  const float* fh_w    = (const float*)d_in[5];
  const float* fh_b    = (const float*)d_in[6];
  const float* iofux_w = (const float*)d_in[7];
  const float* iofux_b = (const float*)d_in[8];
  const float* iofuh_w = (const float*)d_in[9];
  const float* iofuh_b = (const float*)d_in[10];
  float* out = (float*)d_out;

  const int Nn = in_sizes[0] / INDIM;          // number of tree nodes (2^depth - 1)
  int depth = 1;
  while (((1 << depth) - 1) < Nn) ++depth;
  const int maxLvl  = 1 << (depth - 1);        // leaf level width
  const int halfLvl = maxLvl >> 1;
  const int NnPad   = 1 << depth;
  const int NCHUNK  = 32;

  // ---------------- Workspace carve-up ----------------
  // fp32 region first (all block sizes multiples of 512 floats -> aligned).
  float* ws = (float*)d_ws;
  size_t off = 0;
  float* xgbuf   = ws + off; off += (size_t)maxLvl  * 4 * MDIM;  // per-level input gates
  float* h1      = ws + off; off += (size_t)NnPad   * MDIM;
  float* c1      = ws + off; off += (size_t)NnPad   * MDIM;
  float* h2      = ws + off; off += (size_t)NnPad   * MDIM;
  float* c2      = ws + off; off += (size_t)NnPad   * MDIM;
  float* iou     = ws + off; off += (size_t)halfLvl * 3 * MDIM;
  float* fproj   = ws + off; off += (size_t)maxLvl  * MDIM;
  float* proj    = ws + off; off += (size_t)halfLvl * 4 * MDIM;
  float* partial = ws + off; off += (size_t)NCHUNK  * MDIM;
  // f16 region (each buffer size a multiple of 512 elements -> 32B aligned).
  _Float16* f16b = (_Float16*)(ws + off);
  size_t o16 = 0;
  _Float16* in16     = f16b + o16; o16 += (size_t)NnPad   * INDIM;
  _Float16* wfx16    = f16b + o16; o16 += (size_t)4 * MDIM * INDIM;   // ifoux_w
  _Float16* wios16   = f16b + o16; o16 += (size_t)3 * MDIM * MDIM;    // ious_w
  _Float16* wfh16    = f16b + o16; o16 += (size_t)MDIM * MDIM;        // fh_w
  _Float16* wcx16    = f16b + o16; o16 += (size_t)4 * MDIM * INDIM;   // iofux_w
  _Float16* wch16    = f16b + o16; o16 += (size_t)4 * MDIM * MDIM;    // iofuh_w
  _Float16* h1_16    = f16b + o16; o16 += (size_t)NnPad * MDIM;
  _Float16* h2_16    = f16b + o16; o16 += (size_t)NnPad * MDIM;
  _Float16* hsum16   = f16b + o16; o16 += (size_t)halfLvl * MDIM;
  (void)ws_size; (void)n_in; (void)out_size;

  auto cvt = [&](const float* src, _Float16* dst, size_t n) {
    int blocks = (int)((n + 255) / 256); if (blocks > 4096) blocks = 4096;
    cvt_f16_kernel<<<dim3(blocks), 256, 0, stream>>>(src, dst, n);
  };
  auto gemm = [&](const _Float16* A, const _Float16* W, const float* bias, float* C,
                  int M_, int N_, int K_) {
    dim3 grid((M_ + 127) / 128, N_ / 64);
    gemm_nt_wmma_f16<<<grid, dim3(256), 0, stream>>>(A, W, bias, C, M_, N_, K_);
  };
  auto ew_blocks = [](int nl) { return dim3((nl * MDIM + 255) / 256); };

  // ---------------- One-time f16 staging of static operands ----------------
  cvt(inputs,  in16,   (size_t)Nn * INDIM);
  cvt(ifoux_w, wfx16,  (size_t)4 * MDIM * INDIM);
  cvt(ious_w,  wios16, (size_t)3 * MDIM * MDIM);
  cvt(fh_w,    wfh16,  (size_t)MDIM * MDIM);
  cvt(iofux_w, wcx16,  (size_t)4 * MDIM * INDIM);
  cvt(iofuh_w, wch16,  (size_t)4 * MDIM * MDIM);

  // ---------------- Child-sum TreeLSTM: bottom-up ----------------
  for (int lvl = depth - 1; lvl >= 0; --lvl) {
    const int lo = (1 << lvl) - 1;
    const int nl = 1 << lvl;
    // xg = inputs[level] @ ifoux_w.T + ifoux_b   (nl x 2048)
    gemm(in16 + (size_t)lo * INDIM, wfx16, ifoux_b, xgbuf, nl, 4 * MDIM, INDIM);
    if (lvl == depth - 1) {
      cs_leaf_kernel<<<ew_blocks(nl), 256, 0, stream>>>(xgbuf, ious_b, h1, c1, h1_16, lo, nl);
    } else {
      const int clo = 2 * lo + 1;   // children occupy contiguous rows [clo, clo+2*nl)
      cs_hsum_kernel<<<ew_blocks(nl), 256, 0, stream>>>(h1, hsum16, clo, nl);
      gemm(hsum16, wios16, nullptr, iou, nl, 3 * MDIM, MDIM);                      // nl x 1536
      gemm(h1_16 + (size_t)clo * MDIM, wfh16, nullptr, fproj, 2 * nl, MDIM, MDIM); // 2nl x 512
      cs_combine_kernel<<<ew_blocks(nl), 256, 0, stream>>>(
          xgbuf, iou, fproj, ious_b, fh_b, h1, c1, h1_16, lo, clo, nl);
    }
  }

  // ---------------- Chain TreeLSTM: top-down ----------------
  for (int lvl = 0; lvl < depth; ++lvl) {
    const int lo = (1 << lvl) - 1;
    const int nl = 1 << lvl;
    gemm(in16 + (size_t)lo * INDIM, wcx16, iofux_b, xgbuf, nl, 4 * MDIM, INDIM);
    int plo = 0, hasp = 0;
    if (lvl > 0) {
      plo = (1 << (lvl - 1)) - 1;
      const int np = 1 << (lvl - 1);
      // proj = h_parentlevel @ iofuh_w.T   (np x 2048), shared by both children
      gemm(h2_16 + (size_t)plo * MDIM, wch16, nullptr, proj, np, 4 * MDIM, MDIM);
      hasp = 1;
    }
    chain_combine_kernel<<<ew_blocks(nl), 256, 0, stream>>>(
        xgbuf, proj, iofuh_b, h2, c2, h2_16, lo, plo, nl, hasp);
  }

  // ---------------- Output: [h1(root) | colmax(h2)] ----------------
  colmax_partial_kernel<<<dim3(MDIM / 256, NCHUNK), 256, 0, stream>>>(h2, partial, Nn);
  finalize_kernel<<<dim3(MDIM / 256), 256, 0, stream>>>(h1, partial, out, NCHUNK);
}
